// GraphSAGE_80788334838320
// MI455X (gfx1250) — matrix operand order repaired and checked by reading, not compiled, not run on hardware
//
#include <hip/hip_runtime.h>
#include <hip/hip_bf16.h>
#include <stdint.h>

// ---------------- WMMA types ----------------
typedef __attribute__((ext_vector_type(16))) __bf16 v16bf;
typedef __attribute__((ext_vector_type(8)))  float  v8f;

#define HID 64
#define BN_EPS 1e-5f

// f32 -> bf16 round-to-nearest-even via bit ops
static __device__ inline __bf16 bfc(float f) {
    unsigned u = __builtin_bit_cast(unsigned, f);
    unsigned r = u + 0x7FFFu + ((u >> 16) & 1u);
    unsigned short h = (unsigned short)(r >> 16);
    return __builtin_bit_cast(__bf16, h);
}

// -inf (empty segment-max sentinel) -> 0
static __device__ inline float fixneg(float x) {
    return (__builtin_bit_cast(unsigned, x) == 0xFF800000u) ? 0.0f : x;
}

static __device__ inline void packA(v16bf& av, const float4& a0, const float4& a1,
                                    const float4& a2, const float4& a3, float s) {
    av[0]=bfc(a0.x*s);  av[1]=bfc(a0.y*s);  av[2]=bfc(a0.z*s);  av[3]=bfc(a0.w*s);
    av[4]=bfc(a1.x*s);  av[5]=bfc(a1.y*s);  av[6]=bfc(a1.z*s);  av[7]=bfc(a1.w*s);
    av[8]=bfc(a2.x*s);  av[9]=bfc(a2.y*s);  av[10]=bfc(a2.z*s); av[11]=bfc(a2.w*s);
    av[12]=bfc(a3.x*s); av[13]=bfc(a3.y*s); av[14]=bfc(a3.z*s); av[15]=bfc(a3.w*s);
}

// Accumulate A[row,:K] @ W(packed fragments in LDS) into 4 accumulators (16x64 tile).
// A layout (matches packed B): lane<16 -> K {0..15}; lane>=16 -> K {16..31} per ktile.
static __device__ inline void gemm_accum(const float* __restrict__ A,
                                         const int* __restrict__ idx,
                                         const float* __restrict__ rowDiv,
                                         const __bf16* sWbase,  // into LDS staging area
                                         int K, int lda, int fixInf,
                                         int mA, int halfsel, int lane, v8f acc[4]) {
    int row = idx ? idx[mA] : mA;
    float s = rowDiv ? (1.0f / fmaxf(rowDiv[row], 1.0f)) : 1.0f;
    const float* ar = A + (size_t)row * (size_t)lda + (halfsel << 4);
    int nkt = K >> 5;
    for (int kt = 0; kt < nkt; ++kt) {
        const float* a = ar + (kt << 5);
        float4 a0 = *(const float4*)(a);
        float4 a1 = *(const float4*)(a + 4);
        float4 a2 = *(const float4*)(a + 8);
        float4 a3 = *(const float4*)(a + 12);
        if (fixInf) {
            a0.x=fixneg(a0.x); a0.y=fixneg(a0.y); a0.z=fixneg(a0.z); a0.w=fixneg(a0.w);
            a1.x=fixneg(a1.x); a1.y=fixneg(a1.y); a1.z=fixneg(a1.z); a1.w=fixneg(a1.w);
            a2.x=fixneg(a2.x); a2.y=fixneg(a2.y); a2.z=fixneg(a2.z); a2.w=fixneg(a2.w);
            a3.x=fixneg(a3.x); a3.y=fixneg(a3.y); a3.z=fixneg(a3.z); a3.w=fixneg(a3.w);
        }
        v16bf av; packA(av, a0, a1, a2, a3, s);
        const v16bf* wb = (const v16bf*)sWbase + ((size_t)(kt << 2) * 32) + lane;
#pragma unroll
        for (int nt = 0; nt < 4; ++nt) {
            v16bf bv = wb[nt * 32];                 // ds_load from staged fragments
            acc[nt] = __builtin_amdgcn_wmma_f32_16x16x32_bf16(
                false, av, false, bv, (short)0, acc[nt], false, false);
        }
    }
}

// C[M,64] = A1 @ W1 (+ A2 @ W2) + bias, then optional BN / ReLU epilogue.
// mode: 0 = +bias ; 1 = +bias, BN, ReLU ; 2 = +bias, ReLU
__global__ __launch_bounds__(256)
void gemm_h64_kernel(const float* __restrict__ A1, const int* __restrict__ idx1,
                     const float* __restrict__ rowDiv1, int fixInf1,
                     const __bf16* __restrict__ Wp1, int K1, int lda1,
                     const float* __restrict__ A2, const int* __restrict__ idx2,
                     const __bf16* __restrict__ Wp2, int K2, int lda2,
                     const float* __restrict__ bias,
                     const float* __restrict__ gamma, const float* __restrict__ beta,
                     const float* __restrict__ mean,  const float* __restrict__ var,
                     float* __restrict__ out, int M, int mode) {
    __shared__ alignas(32) __bf16 sW[16384];   // 32KB: covers K1+K2 <= 256 columns of 64

    // ---- stage packed weights (Wp1 then Wp2, contiguous) into LDS via async DMA ----
    {
        const int k1Bytes  = K1 * HID * 2;
        const int totBytes = (K1 + K2) * HID * 2;
        const int nChunk   = totBytes >> 4;                 // 16B chunks
        unsigned sBase = (unsigned)(size_t)(&sW[0]);
        for (int i = threadIdx.x; i < nChunk; i += 256) {
            int boff = i << 4;
            const char* g = (boff < k1Bytes) ? ((const char*)Wp1 + boff)
                                             : ((const char*)Wp2 + (boff - k1Bytes));
            unsigned l = sBase + (unsigned)boff;
            asm volatile("global_load_async_to_lds_b128 %0, %1, off"
                         :: "v"(l), "v"(g) : "memory");
        }
        asm volatile("s_wait_asynccnt 0x0" ::: "memory");
        __syncthreads();
    }

    int wave = threadIdx.x >> 5;
    int lane = threadIdx.x & 31;
    int tileRow = (blockIdx.x * 8 + wave) << 4;
    if (tileRow >= M) return;                  // wave-uniform exit (after barrier)
    int halfsel = lane >> 4;                   // 0/1
    int rA = lane & 15;
    int mA = tileRow + rA; if (mA >= M) mA = M - 1;

    v8f acc[4];
#pragma unroll
    for (int t = 0; t < 4; ++t) acc[t] = (v8f){0,0,0,0,0,0,0,0};

    gemm_accum(A1, idx1, rowDiv1, &sW[0], K1, lda1, fixInf1, mA, halfsel, lane, acc);
    if (A2) gemm_accum(A2, idx2, nullptr, &sW[(size_t)K1 * HID], K2, lda2, 0,
                       mA, halfsel, lane, acc);

    int rowOffs = halfsel << 3;
#pragma unroll
    for (int nt = 0; nt < 4; ++nt) {
        int n = (nt << 4) + rA;
        float b = bias[n];
        float sc = 1.0f, sh = b;
        if (mode == 1) {
            sc = gamma[n] * rsqrtf(var[n] + BN_EPS);
            sh = (b - mean[n]) * sc + beta[n];
        }
#pragma unroll
        for (int r = 0; r < 8; ++r) {
            int mm = tileRow + rowOffs + r;
            if (mm < M) {
                float c = acc[nt][r];
                float y = (mode == 1) ? (c * sc + sh) : (c + b);
                if (mode != 0) y = fmaxf(y, 0.0f);
                out[(size_t)mm * HID + n] = y;
            }
        }
    }
}

// Pack W[K,64] (f32 row-major) into per-wmma B-fragment layout:
// [(kt*4+nt)*32 + lane][16 bf16], lane<16: n=lane, k=kt*32+j ; lane>=16: n=lane-16, k=kt*32+16+j
__global__ void pack_w_kernel(const float* __restrict__ W, __bf16* __restrict__ Wp, int K) {
    int tid = blockIdx.x * blockDim.x + threadIdx.x;
    if (tid >= K * HID) return;
    int j    = tid & 15;
    int lane = (tid >> 4) & 31;
    int ntkt = tid >> 9;
    int nt   = ntkt & 3;
    int kt   = ntkt >> 2;
    int n = (nt << 4) + (lane & 15);
    int k = (kt << 5) + ((lane >> 4) << 4) + j;
    Wp[tid] = bfc(W[(size_t)k * HID + n]);
}

// ---------------- fills ----------------
__global__ void fill_u32_kernel(unsigned* __restrict__ p, unsigned v, long long n) {
    long long i = (long long)blockIdx.x * blockDim.x + threadIdx.x;
    if (i < n) p[i] = v;
}

// ---------------- edge aggregation ----------------
__global__ void deg_kernel(const int* __restrict__ edst, float* __restrict__ deg, int E) {
    int e = blockIdx.x * blockDim.x + threadIdx.x;
    if (e < E) atomicAdd(&deg[edst[e]], 1.0f);
}

static __device__ inline void atomicMaxF(float* addr, float val) {
    if (val >= 0.0f) atomicMax((int*)addr, __float_as_int(val));
    else             atomicMin((unsigned int*)addr, __float_as_uint(val));
}

// agg_u[src[e],:] = max(agg_u[src[e],:], ix[dst[e],:]) ; 16 threads per edge, float4 each
__global__ void scatter_max_kernel(const int* __restrict__ esrc, const int* __restrict__ edst,
                                   const float* __restrict__ ix, float* __restrict__ agg,
                                   long long total) {
    long long t = (long long)blockIdx.x * blockDim.x + threadIdx.x;
    if (t >= total) return;
    int e = (int)(t >> 4);
    int c = (int)(t & 15) << 2;
    int u  = esrc[e];
    int it = edst[e];
    float4 v = *(const float4*)(ix + (size_t)it * HID + c);
    float* a = agg + (size_t)u * HID + c;
    atomicMaxF(a + 0, v.x); atomicMaxF(a + 1, v.y);
    atomicMaxF(a + 2, v.z); atomicMaxF(a + 3, v.w);
}

// agg_i[dst[e],:] += ux[src[e],:]
__global__ void scatter_add_kernel(const int* __restrict__ esrc, const int* __restrict__ edst,
                                   const float* __restrict__ ux, float* __restrict__ agg,
                                   long long total) {
    long long t = (long long)blockIdx.x * blockDim.x + threadIdx.x;
    if (t >= total) return;
    int e = (int)(t >> 4);
    int c = (int)(t & 15) << 2;
    int u  = esrc[e];
    int it = edst[e];
    float4 v = *(const float4*)(ux + (size_t)u * HID + c);
    float* a = agg + (size_t)it * HID + c;
    atomicAdd(a + 0, v.x); atomicAdd(a + 1, v.y);
    atomicAdd(a + 2, v.z); atomicAdd(a + 3, v.w);
}

// ---------------- final fc2: out[m,0:4] = hid[m,:] @ W[64,4] + b ----------------
__global__ void fc2_kernel(const float* __restrict__ hid, const float* __restrict__ W,
                           const float* __restrict__ b, float* __restrict__ out, int M) {
    int m = blockIdx.x * blockDim.x + threadIdx.x;
    if (m >= M) return;
    float a0 = b[0], a1 = b[1], a2 = b[2], a3 = b[3];
    const float* h = hid + (size_t)m * HID;
#pragma unroll 8
    for (int k = 0; k < HID; ++k) {
        float x = h[k];
        const float* w = W + k * 4;
        a0 += x * w[0]; a1 += x * w[1]; a2 += x * w[2]; a3 += x * w[3];
    }
    float4 r = make_float4(a0, a1, a2, a3);
    *(float4*)(out + (size_t)m * 4) = r;
}

// ---------------- host-side orchestration ----------------
extern "C" void kernel_launch(void* const* d_in, const int* in_sizes, int n_in,
                              void* d_out, int out_size, void* d_ws, size_t ws_size,
                              hipStream_t stream) {
    if (n_in < 26) return;
    const int NU = in_sizes[0] / 256;   // 100000
    const int NI = in_sizes[1] / 128;   // 50000
    const int E  = in_sizes[2] / 2;     // 2000000
    const int B  = in_sizes[3] / 2;     // 100000
    const int L  = 2;

    const float* user_feat = (const float*)d_in[0];
    const float* item_feat = (const float*)d_in[1];
    const int*   esrc      = (const int*)d_in[2];          // [E] user idx
    const int*   edst      = ((const int*)d_in[2]) + E;    // [E] item idx
    const int*   lbl_u     = (const int*)d_in[3];          // [B]
    const int*   lbl_i     = ((const int*)d_in[3]) + B;    // [B]
    const float* uproj_W = (const float*)d_in[4];
    const float* uproj_b = (const float*)d_in[5];
    const float* iproj_W = (const float*)d_in[6];
    const float* iproj_b = (const float*)d_in[7];
    const float* u_ll_W  = (const float*)d_in[8];   // [L,64,64]
    const float* u_ll_b  = (const float*)d_in[9];   // [L,64]
    const float* u_lr_W  = (const float*)d_in[10];
    const float* i_ll_W  = (const float*)d_in[11];
    const float* i_ll_b  = (const float*)d_in[12];
    const float* i_lr_W  = (const float*)d_in[13];
    const float* u_g = (const float*)d_in[14];
    const float* u_b = (const float*)d_in[15];
    const float* u_m = (const float*)d_in[16];
    const float* u_v = (const float*)d_in[17];
    const float* i_g = (const float*)d_in[18];
    const float* i_b = (const float*)d_in[19];
    const float* i_m = (const float*)d_in[20];
    const float* i_v = (const float*)d_in[21];
    const float* fc1_W = (const float*)d_in[22];    // [128,64]
    const float* fc1_b = (const float*)d_in[23];
    const float* fc2_W = (const float*)d_in[24];    // [64,4]
    const float* fc2_b = (const float*)d_in[25];
    float* out = (float*)d_out;
    (void)out_size;

    // workspace carve-up (256B aligned)
    size_t off = 0;
    auto carve = [&](size_t bytes) -> void* {
        void* p = (char*)d_ws + off;
        off += (bytes + 255) & ~(size_t)255;
        return p;
    };
    float* ux0   = (float*)carve((size_t)NU * HID * 4);
    float* ux1   = (float*)carve((size_t)NU * HID * 4);
    float* ix0   = (float*)carve((size_t)NI * HID * 4);
    float* ix1   = (float*)carve((size_t)NI * HID * 4);
    float* agg_u = (float*)carve((size_t)NU * HID * 4);   // reused as fc1 hidden
    float* agg_i = (float*)carve((size_t)NI * HID * 4);
    float* deg   = (float*)carve((size_t)NI * 4);
    __bf16* pw_uproj = (__bf16*)carve((size_t)256 * HID * 2);
    __bf16* pw_iproj = (__bf16*)carve((size_t)128 * HID * 2);
    __bf16* pw_ull[2], *pw_ulr[2], *pw_ill[2], *pw_ilr[2];
    for (int l = 0; l < L; ++l) {
        pw_ull[l] = (__bf16*)carve((size_t)64 * HID * 2);
        pw_ulr[l] = (__bf16*)carve((size_t)64 * HID * 2);
        pw_ill[l] = (__bf16*)carve((size_t)64 * HID * 2);
        pw_ilr[l] = (__bf16*)carve((size_t)64 * HID * 2);
    }
    __bf16* pw_fc1 = (__bf16*)carve((size_t)128 * HID * 2);   // kt-major: first 2 ktiles = rows 0..63
    if (off > ws_size) return;

    const int TB = 256;
    auto grid1 = [](long long n, int tb) { return (unsigned)((n + tb - 1) / tb); };

    // ---- pack weights into WMMA B-fragment layout ----
    hipLaunchKernelGGL(pack_w_kernel, dim3(grid1(256 * HID, TB)), dim3(TB), 0, stream, uproj_W, pw_uproj, 256);
    hipLaunchKernelGGL(pack_w_kernel, dim3(grid1(128 * HID, TB)), dim3(TB), 0, stream, iproj_W, pw_iproj, 128);
    for (int l = 0; l < L; ++l) {
        hipLaunchKernelGGL(pack_w_kernel, dim3(grid1(64 * HID, TB)), dim3(TB), 0, stream, u_ll_W + (size_t)l * 64 * 64, pw_ull[l], 64);
        hipLaunchKernelGGL(pack_w_kernel, dim3(grid1(64 * HID, TB)), dim3(TB), 0, stream, u_lr_W + (size_t)l * 64 * 64, pw_ulr[l], 64);
        hipLaunchKernelGGL(pack_w_kernel, dim3(grid1(64 * HID, TB)), dim3(TB), 0, stream, i_ll_W + (size_t)l * 64 * 64, pw_ill[l], 64);
        hipLaunchKernelGGL(pack_w_kernel, dim3(grid1(64 * HID, TB)), dim3(TB), 0, stream, i_lr_W + (size_t)l * 64 * 64, pw_ilr[l], 64);
    }
    hipLaunchKernelGGL(pack_w_kernel, dim3(grid1(128 * HID, TB)), dim3(TB), 0, stream, fc1_W, pw_fc1, 128);

    // ---- degrees ----
    hipLaunchKernelGGL(fill_u32_kernel, dim3(grid1(NI, TB)), dim3(TB), 0, stream, (unsigned*)deg, 0u, (long long)NI);
    hipLaunchKernelGGL(deg_kernel, dim3(grid1(E, TB)), dim3(TB), 0, stream, edst, deg, E);

    // ---- projections: ux0 = user_feat @ uproj + b ; ix0 = item_feat @ iproj + b ----
    hipLaunchKernelGGL(gemm_h64_kernel, dim3((NU + 127) / 128), dim3(TB), 0, stream,
                       user_feat, (const int*)nullptr, (const float*)nullptr, 0, pw_uproj, 256, 256,
                       (const float*)nullptr, (const int*)nullptr, (const __bf16*)nullptr, 0, 0,
                       uproj_b, (const float*)nullptr, (const float*)nullptr, (const float*)nullptr, (const float*)nullptr,
                       ux0, NU, 0);
    hipLaunchKernelGGL(gemm_h64_kernel, dim3((NI + 127) / 128), dim3(TB), 0, stream,
                       item_feat, (const int*)nullptr, (const float*)nullptr, 0, pw_iproj, 128, 128,
                       (const float*)nullptr, (const int*)nullptr, (const __bf16*)nullptr, 0, 0,
                       iproj_b, (const float*)nullptr, (const float*)nullptr, (const float*)nullptr, (const float*)nullptr,
                       ix0, NI, 0);

    // ---- layers ----
    float* uxOld = ux0; float* uxNew = ux1;
    float* ixOld = ix0; float* ixNew = ix1;
    const long long e16 = (long long)E * 16;
    for (int l = 0; l < L; ++l) {
        // user conv: segment_max over item neighbors (-inf fixup folded into GEMM A-load)
        hipLaunchKernelGGL(fill_u32_kernel, dim3(grid1((long long)NU * HID, TB)), dim3(TB), 0, stream,
                           (unsigned*)agg_u, 0xFF800000u, (long long)NU * HID);
        hipLaunchKernelGGL(scatter_max_kernel, dim3(grid1(e16, TB)), dim3(TB), 0, stream,
                           esrc, edst, ixOld, agg_u, e16);
        hipLaunchKernelGGL(gemm_h64_kernel, dim3((NU + 127) / 128), dim3(TB), 0, stream,
                           agg_u, (const int*)nullptr, (const float*)nullptr, 1, pw_ull[l], 64, HID,
                           uxOld, (const int*)nullptr, pw_ulr[l], 64, HID,
                           u_ll_b + (size_t)l * HID, u_g + (size_t)l * HID, u_b + (size_t)l * HID,
                           u_m + (size_t)l * HID, u_v + (size_t)l * HID,
                           uxNew, NU, 1);
        // item conv: segment_mean over user neighbors (uses OLD ux; deg divide folded into GEMM)
        hipLaunchKernelGGL(fill_u32_kernel, dim3(grid1((long long)NI * HID, TB)), dim3(TB), 0, stream,
                           (unsigned*)agg_i, 0u, (long long)NI * HID);
        hipLaunchKernelGGL(scatter_add_kernel, dim3(grid1(e16, TB)), dim3(TB), 0, stream,
                           esrc, edst, uxOld, agg_i, e16);
        hipLaunchKernelGGL(gemm_h64_kernel, dim3((NI + 127) / 128), dim3(TB), 0, stream,
                           agg_i, (const int*)nullptr, deg, 0, pw_ill[l], 64, HID,
                           ixOld, (const int*)nullptr, pw_ilr[l], 64, HID,
                           i_ll_b + (size_t)l * HID, i_g + (size_t)l * HID, i_b + (size_t)l * HID,
                           i_m + (size_t)l * HID, i_v + (size_t)l * HID,
                           ixNew, NI, 1);
        // swap
        float* t;
        t = uxOld; uxOld = uxNew; uxNew = t;
        t = ixOld; ixOld = ixNew; ixNew = t;
    }

    // ---- head: hid = relu([ux[lu], ix[li]] @ fc1 + b) via dual gathered GEMM ----
    float* hid = agg_u;  // reuse
    hipLaunchKernelGGL(gemm_h64_kernel, dim3((B + 127) / 128), dim3(TB), 0, stream,
                       uxOld, lbl_u, (const float*)nullptr, 0, pw_fc1, 64, HID,
                       ixOld, lbl_i, pw_fc1 + (size_t)2 * 4 * 32 * 16, 64, HID,
                       fc1_b, (const float*)nullptr, (const float*)nullptr, (const float*)nullptr, (const float*)nullptr,
                       hid, B, 2);
    // ---- fc2 ----
    hipLaunchKernelGGL(fc2_kernel, dim3(grid1(B, TB)), dim3(TB), 0, stream, hid, fc2_W, fc2_b, out, B);
}
